// AG_MOE_RS_36816459661329
// MI455X (gfx1250) — matrix operand order repaired and checked by reading, not compiled, not run on hardware
//
#include <hip/hip_runtime.h>
#include <hip/hip_bf16.h>

// ---------------- problem constants (from reference setup_inputs) ------------
constexpr int T  = 4096;
constexpr int H  = 1024;
constexpr int I  = 1024;
constexpr int E  = 8;
constexpr int K  = 2;
constexpr int TK = T * K;          // 8192 token-slots

// ---------------- vector types for WMMA --------------------------------------
typedef __bf16 v16bf __attribute__((ext_vector_type(16)));
typedef __bf16 v8bf  __attribute__((ext_vector_type(8)));
typedef float  v8f   __attribute__((ext_vector_type(8)));

static __device__ __forceinline__ __bf16 f32_to_bf16(float f) {
    unsigned u = __builtin_bit_cast(unsigned, f);
    unsigned r = (u + 0x7FFFu + ((u >> 16) & 1u)) >> 16;
    unsigned short s = (unsigned short)r;
    return __builtin_bit_cast(__bf16, s);
}

static __device__ __forceinline__ v16bf cat8(v8bf lo, v8bf hi) {
    return __builtin_shufflevector(lo, hi, 0, 1, 2, 3, 4, 5, 6, 7,
                                   8, 9, 10, 11, 12, 13, 14, 15);
}

// Async copy: 16 bytes global -> LDS, tracked by ASYNCcnt (CDNA5).
static __device__ __forceinline__ void async_g2l_b128(unsigned lds_off,
                                                      const void* gptr) {
    unsigned long long ga = (unsigned long long)(size_t)gptr;
    asm volatile("global_load_async_to_lds_b128 %0, %1, off"
                 :: "v"(lds_off), "v"(ga) : "memory");
}
static __device__ __forceinline__ void wait_asynccnt0() {
    asm volatile("s_wait_asynccnt 0x0" ::: "memory");
}

// ---------------- workspace layout (bytes, all 256-aligned) ------------------
constexpr size_t O_CNT  = 0;                                   // E int32 (pad 256)
constexpr size_t O_LIST = 256;                                 // E*TK int32
constexpr size_t O_W    = O_LIST + (size_t)E * TK * 4;         // TK f32
constexpr size_t O_X    = O_W + (size_t)TK * 4;                // T*H bf16
constexpr size_t O_UPW  = O_X + (size_t)T * H * 2;             // E*2I*H bf16 (transposed)
constexpr size_t O_DWN  = O_UPW + (size_t)E * 2 * I * H * 2;   // E*H*I bf16 (transposed)
constexpr size_t O_HBUF = O_DWN + (size_t)E * I * H * 2;       // TK*I bf16

// ============================================================================
// 1) Router: softmax over E=8 logits, top-2, renormalize, build expert lists.
// ============================================================================
__global__ void moe_router_kernel(const float* __restrict__ logits,
                                  int* __restrict__ cnt,
                                  int* __restrict__ lists,
                                  float* __restrict__ wts) {
    int t = blockIdx.x * blockDim.x + threadIdx.x;
    if (t >= T) return;
    float l[E];
    float mx = -1e30f;
    #pragma unroll
    for (int e = 0; e < E; ++e) { l[e] = logits[t * E + e]; mx = fmaxf(mx, l[e]); }
    #pragma unroll
    for (int e = 0; e < E; ++e) l[e] = __expf(l[e] - mx);   // softmax denom cancels after renorm
    int i0 = 0;
    #pragma unroll
    for (int e = 1; e < E; ++e) if (l[e] > l[i0]) i0 = e;
    int i1 = (i0 == 0) ? 1 : 0;
    #pragma unroll
    for (int e = 0; e < E; ++e) if (e != i0 && l[e] > l[i1]) i1 = e;
    float p0 = l[i0], p1 = l[i1];
    float inv = 1.0f / (p0 + p1);
    wts[t * K + 0] = p0 * inv;
    wts[t * K + 1] = p1 * inv;
    int pos0 = atomicAdd(&cnt[i0], 1);
    lists[i0 * TK + pos0] = t * K + 0;
    int pos1 = atomicAdd(&cnt[i1], 1);
    lists[i1 * TK + pos1] = t * K + 1;
}

// ============================================================================
// 2) f32 -> bf16 converts (x elementwise; weights transposed to N-major so the
//    WMMA B-fragment is contiguous-per-lane).
// ============================================================================
__global__ void convert_x_kernel(const float* __restrict__ in, __bf16* __restrict__ out, int n) {
    int i = blockIdx.x * blockDim.x + threadIdx.x;
    if (i < n) out[i] = f32_to_bf16(in[i]);
}

// up_weight [E, H, 2I] -> [E, 2I, H]
__global__ void transpose_up_kernel(const float* __restrict__ in, __bf16* __restrict__ out) {
    size_t idx = (size_t)blockIdx.x * blockDim.x + threadIdx.x;
    if (idx >= (size_t)E * H * 2 * I) return;
    int e = (int)(idx / ((size_t)H * 2 * I));
    int r = (int)(idx % ((size_t)H * 2 * I));
    int k = r / (2 * I);
    int n = r % (2 * I);
    out[((size_t)e * 2 * I + n) * H + k] = f32_to_bf16(in[idx]);
}

// down_weight [E, I, H] -> [E, H, I]
__global__ void transpose_down_kernel(const float* __restrict__ in, __bf16* __restrict__ out) {
    size_t idx = (size_t)blockIdx.x * blockDim.x + threadIdx.x;
    if (idx >= (size_t)E * I * H) return;
    int e = (int)(idx / ((size_t)I * H));
    int r = (int)(idx % ((size_t)I * H));
    int i = r / H;
    int hc = r % H;
    out[((size_t)e * H + hc) * I + i] = f32_to_bf16(in[idx]);
}

// ============================================================================
// 3) Up-projection grouped GEMM + gated SiLU, bf16 WMMA, f32 accum.
//    Block: 8 waves. Tile: 32 slots x 128 gate cols (+128 paired up cols).
//    A tile staged via async global->LDS (ASYNCcnt), 4 WMMAs/wave/k-step.
// ============================================================================
__global__ void __launch_bounds__(256)
moe_up_gemm_kernel(const __bf16* __restrict__ xb,
                   const __bf16* __restrict__ upw,   // [E, 2I, H]
                   const int* __restrict__ cnt,
                   const int* __restrict__ lists,
                   __bf16* __restrict__ hbuf) {      // [TK, I]
    const int e = blockIdx.z;
    const int tile0 = blockIdx.y * 32;
    const int ce = cnt[e];
    if (tile0 >= ce) return;

    __shared__ __align__(16) __bf16 shA[32 * 32];   // 32 slots x 32 k, 2 KB
    __shared__ int shSlot[32];

    const int tid  = threadIdx.x;
    const int lane = tid & 31;
    const int wave = tid >> 5;

    if (tid < 32) {
        int idx = tile0 + tid;
        shSlot[tid] = (idx < ce) ? lists[e * TK + idx] : -1;
    }
    __syncthreads();

    // ---- staging config: threads 0..127 move one 16B chunk per k-step ------
    const bool stager = tid < 128;                   // waves 0..3
    const int  srow = tid >> 2;                      // 0..31
    const int  sc   = (tid & 3) * 8;                 // 0,8,16,24 (bf16 elems)
    bool svalid = false;
    const __bf16* sptr = xb;
    unsigned ldsoff = 0;
    if (stager) {
        int s = shSlot[srow];
        svalid = (s >= 0);
        int tok = (s >= 0 ? s : 0) / K;
        sptr = xb + (size_t)tok * H + sc;
        ldsoff = (unsigned)(size_t)(&shA[srow * 32 + sc]);
        if (!svalid) {                               // zero dead rows once
            v8bf z = {};
            *(v8bf*)(&shA[srow * 32 + sc]) = z;
        }
    }

    const int ncol = blockIdx.x * 128 + wave * 16 + (lane & 15); // gate col in [0,I)
    const int khi  = 8 * (lane >> 4);    // B sub-k offset
    const int akb  = 8 * (lane >> 4);    // A sub-k offset
    const int am   = lane & 15;          // A row within fragment

    const __bf16* bg = upw + ((size_t)e * 2 * I + ncol) * H;       // gate B row
    const __bf16* bu = upw + ((size_t)e * 2 * I + ncol + I) * H;   // up   B row

    v8f acc0g = {}, acc0u = {}, acc1g = {}, acc1u = {};

    for (int k0 = 0; k0 < H; k0 += 32) {
        __syncthreads();                 // previous consumers done (also covers init)
        if (stager) {
            if (svalid) {
                async_g2l_b128(ldsoff, sptr);
                sptr += 32;
            }
            wait_asynccnt0();
        }
        __syncthreads();                 // LDS tile visible to all waves

        // A fragments (two 16-row halves of the 32-slot tile)
        const v8bf* ap0 = (const v8bf*)(shA + am * 32 + akb);
        const v8bf* ap1 = (const v8bf*)(shA + (16 + am) * 32 + akb);
        v16bf a0 = cat8(ap0[0], ap0[2]);
        v16bf a1 = cat8(ap1[0], ap1[2]);
        // B fragments: col ncol, K = k0+khi+{0..7} and k0+16+khi+{0..7}
        const v8bf* bgp = (const v8bf*)(bg + k0 + khi);
        const v8bf* bup = (const v8bf*)(bu + k0 + khi);
        v16bf b_g = cat8(bgp[0], bgp[2]);
        v16bf b_u = cat8(bup[0], bup[2]);
        __builtin_prefetch(bg + k0 + 128, 0, 1);
        __builtin_prefetch(bu + k0 + 128, 0, 1);

        acc0g = __builtin_amdgcn_wmma_f32_16x16x32_bf16(false, a0, false, b_g,
                                                        (short)0, acc0g, false, false);
        acc0u = __builtin_amdgcn_wmma_f32_16x16x32_bf16(false, a0, false, b_u,
                                                        (short)0, acc0u, false, false);
        acc1g = __builtin_amdgcn_wmma_f32_16x16x32_bf16(false, a1, false, b_g,
                                                        (short)0, acc1g, false, false);
        acc1u = __builtin_amdgcn_wmma_f32_16x16x32_bf16(false, a1, false, b_u,
                                                        (short)0, acc1u, false, false);
    }

    // gated SiLU on f32 accumulators, store bf16 h per slot
    #pragma unroll
    for (int f = 0; f < 2; ++f) {
        #pragma unroll
        for (int r = 0; r < 8; ++r) {
            int m = f * 16 + r + 8 * (lane >> 4);
            int s = shSlot[m];
            if (s >= 0) {
                float g = (f == 0) ? acc0g[r] : acc1g[r];
                float u = (f == 0) ? acc0u[r] : acc1u[r];
                float hv = (g / (1.0f + __expf(-g))) * u;
                hbuf[(size_t)s * I + ncol] = f32_to_bf16(hv);
            }
        }
    }
}

// ============================================================================
// 4) Down-projection grouped GEMM + weighted combine (atomic f32 scatter-add).
//    Tile: 32 slots x 128 H cols, async A staging, 2 WMMAs/wave/k-step.
// ============================================================================
__global__ void __launch_bounds__(256)
moe_down_gemm_kernel(const __bf16* __restrict__ hbuf,  // [TK, I]
                     const __bf16* __restrict__ dwn,   // [E, H, I]
                     const int* __restrict__ cnt,
                     const int* __restrict__ lists,
                     const float* __restrict__ wts,    // [TK]
                     float* __restrict__ out) {        // [T, H]
    const int e = blockIdx.z;
    const int tile0 = blockIdx.y * 32;
    const int ce = cnt[e];
    if (tile0 >= ce) return;

    __shared__ __align__(16) __bf16 shA[32 * 32];
    __shared__ int shSlot[32];
    __shared__ float shW[32];

    const int tid  = threadIdx.x;
    const int lane = tid & 31;
    const int wave = tid >> 5;

    if (tid < 32) {
        int idx = tile0 + tid;
        int s = (idx < ce) ? lists[e * TK + idx] : -1;
        shSlot[tid] = s;
        shW[tid] = (s >= 0) ? wts[s] : 0.0f;
    }
    __syncthreads();

    const bool stager = tid < 128;
    const int  srow = tid >> 2;
    const int  sc   = (tid & 3) * 8;
    bool svalid = false;
    const __bf16* sptr = hbuf;
    unsigned ldsoff = 0;
    if (stager) {
        int s = shSlot[srow];
        svalid = (s >= 0);
        sptr = hbuf + (size_t)(s >= 0 ? s : 0) * I + sc;
        ldsoff = (unsigned)(size_t)(&shA[srow * 32 + sc]);
        if (!svalid) {
            v8bf z = {};
            *(v8bf*)(&shA[srow * 32 + sc]) = z;
        }
    }

    const int hc  = blockIdx.x * 128 + wave * 16 + (lane & 15);  // output col
    const int khi = 8 * (lane >> 4);
    const int akb = 8 * (lane >> 4);
    const int am  = lane & 15;

    const __bf16* bp = dwn + ((size_t)e * H + hc) * I;

    v8f acc0 = {}, acc1 = {};

    for (int i0 = 0; i0 < I; i0 += 32) {
        __syncthreads();
        if (stager) {
            if (svalid) {
                async_g2l_b128(ldsoff, sptr);
                sptr += 32;
            }
            wait_asynccnt0();
        }
        __syncthreads();

        const v8bf* ap0 = (const v8bf*)(shA + am * 32 + akb);
        const v8bf* ap1 = (const v8bf*)(shA + (16 + am) * 32 + akb);
        v16bf a0 = cat8(ap0[0], ap0[2]);
        v16bf a1 = cat8(ap1[0], ap1[2]);
        const v8bf* bpp = (const v8bf*)(bp + i0 + khi);
        v16bf b = cat8(bpp[0], bpp[2]);
        __builtin_prefetch(bp + i0 + 128, 0, 1);

        acc0 = __builtin_amdgcn_wmma_f32_16x16x32_bf16(false, a0, false, b,
                                                       (short)0, acc0, false, false);
        acc1 = __builtin_amdgcn_wmma_f32_16x16x32_bf16(false, a1, false, b,
                                                       (short)0, acc1, false, false);
    }

    #pragma unroll
    for (int f = 0; f < 2; ++f) {
        #pragma unroll
        for (int r = 0; r < 8; ++r) {
            int m = f * 16 + r + 8 * (lane >> 4);
            int s = shSlot[m];
            if (s >= 0) {
                float v = ((f == 0) ? acc0[r] : acc1[r]) * shW[m];
                int tok = s / K;
                __hip_atomic_fetch_add(&out[(size_t)tok * H + hc], v,
                                       __ATOMIC_RELAXED, __HIP_MEMORY_SCOPE_AGENT);
            }
        }
    }
}

// ============================================================================
// Launch
// ============================================================================
extern "C" void kernel_launch(void* const* d_in, const int* in_sizes, int n_in,
                              void* d_out, int out_size, void* d_ws, size_t ws_size,
                              hipStream_t stream) {
    (void)in_sizes; (void)n_in; (void)ws_size;
    const float* x      = (const float*)d_in[0];   // [T, H]
    const float* logits = (const float*)d_in[1];   // [T, E]
    const float* upw    = (const float*)d_in[2];   // [E, H, 2I]
    const float* dww    = (const float*)d_in[3];   // [E, I, H]
    // d_in[4] = topk scalar (== 2, compiled in)

    char* ws = (char*)d_ws;
    int*    cnt   = (int*)(ws + O_CNT);
    int*    lists = (int*)(ws + O_LIST);
    float*  wts   = (float*)(ws + O_W);
    __bf16* xb    = (__bf16*)(ws + O_X);
    __bf16* upb   = (__bf16*)(ws + O_UPW);
    __bf16* dwb   = (__bf16*)(ws + O_DWN);
    __bf16* hb    = (__bf16*)(ws + O_HBUF);
    float*  out   = (float*)d_out;

    hipMemsetAsync(cnt, 0, 256, stream);
    hipMemsetAsync(out, 0, (size_t)out_size * sizeof(float), stream);

    moe_router_kernel<<<T / 256, 256, 0, stream>>>(logits, cnt, lists, wts);
    convert_x_kernel<<<(T * H) / 256, 256, 0, stream>>>(x, xb, T * H);
    transpose_up_kernel<<<(unsigned)(((size_t)E * H * 2 * I) / 256), 256, 0, stream>>>(upw, upb);
    transpose_down_kernel<<<(unsigned)(((size_t)E * I * H) / 256), 256, 0, stream>>>(dww, dwb);

    dim3 gUp(I / 128, TK / 32, E);
    moe_up_gemm_kernel<<<gUp, 256, 0, stream>>>(xb, upb, cnt, lists, hb);

    dim3 gDn(H / 128, TK / 32, E);
    moe_down_gemm_kernel<<<gDn, 256, 0, stream>>>(hb, dwb, cnt, lists, wts, out);
}
